// BidirectionalCrossAttentionGate_44126493999715
// MI455X (gfx1250) — compile-verified, matrix operand order checked by reading, not run on hardware
//
#include <hip/hip_runtime.h>
#include <hip/hip_bf16.h>
#include <math.h>

typedef __attribute__((ext_vector_type(16))) _Float16 v16h;
typedef __attribute__((ext_vector_type(8)))  _Float16 v8h;
typedef __attribute__((ext_vector_type(8)))  float    v8f;
typedef __attribute__((ext_vector_type(4)))  int      v4i_;

// Problem constants
#define BN_C 8
#define S_C  2048
#define D_C  768

// ---------------------------------------------------------------------------
// CDNA5 async global->LDS staging helpers (ASYNCcnt-tracked DMA, ISA ch.10).
// Builtin signature (probe-confirmed via diagnostic): (v4i AS1*, v4i AS3*,
// imm offset, imm cpol).
// ---------------------------------------------------------------------------
typedef __attribute__((address_space(1))) v4i_ gv4i;
typedef __attribute__((address_space(3))) v4i_ lv4i;

__device__ __forceinline__ void async_ld_b128(const _Float16* g, _Float16* l) {
#if __has_builtin(__builtin_amdgcn_global_load_async_to_lds_b128)
  __builtin_amdgcn_global_load_async_to_lds_b128(
      (gv4i*)(uintptr_t)g,
      (lv4i*)(unsigned)(uintptr_t)l, 0, 0);
#else
  // VDST = per-lane LDS byte offset, VADDR = 64-bit global address (GV mode)
  asm volatile("global_load_async_to_lds_b128 %0, %1, off"
               :: "v"((unsigned)(uintptr_t)l), "v"(g)
               : "memory");
#endif
}

__device__ __forceinline__ void wait_async0() {
#if __has_builtin(__builtin_amdgcn_s_wait_asynccnt)
  __builtin_amdgcn_s_wait_asynccnt(0);
#else
  asm volatile("s_wait_asynccnt 0x0" ::: "memory");
#endif
}

// ---------------------------------------------------------------------------
// f32 -> f16 conversion
// ---------------------------------------------------------------------------
__global__ void cvt_f32_f16_kernel(const float* __restrict__ src,
                                   _Float16* __restrict__ dst, long n) {
  long i = (long)blockIdx.x * blockDim.x + threadIdx.x;
  if (i < n) dst[i] = (_Float16)src[i];
}

// ---------------------------------------------------------------------------
// Fragment load: one 16-row x 32-k f16 tile row slice (LDS stride 40 halves).
// ISA 7.12.2 16-bit A layout: lanes 0-15 hold rows; k 0-7/8-15 split by lane
// half in VGPRs 0-3, k 16-23/24-31 in VGPRs 4-7 -> two 16B LDS loads/lane.
// ---------------------------------------------------------------------------
__device__ inline v16h load_frag16(const _Float16* base, int half) {
  v8h lo = *(const v8h*)(base + half * 8);
  v8h hi = *(const v8h*)(base + half * 8 + 16);
  v16h f;
#pragma unroll
  for (int i = 0; i < 8; ++i) { f[i] = lo[i]; f[i + 8] = hi[i]; }
  return f;
}

// ---------------------------------------------------------------------------
// 128x64-block WMMA GEMM, C = act(A @ B [+bias]) with optional B^T sourcing.
//   MODE 0: f16 out = acc + bias[col]          (projections)
//   MODE 1: f32 out = acc * scale              (attention logits)
//   MODE 2: f32 out = sigmoid(tanh(acc))       (attention gate)
//   MODE 3: f32 out = acc + bias[col]          (final linear)
// 256 threads = 8 wave32. Each wave owns a 32x32 output patch (2x2 tiles of
// 16x16): 2 A-frags x 2 B-frags -> 4 WMMAs per 32-k step.
// A tile (and B tile when TRANSB) staged by async global->LDS DMA into a
// double buffer; one s_wait_asynccnt + one barrier per iteration. The
// transpose-scatter B path (!TRANSB) keeps register staging, with the LDS
// scatter placed after the WMMAs to hide the load latency.
// ---------------------------------------------------------------------------
template <int MODE, bool TRANSB>
__global__ __launch_bounds__(256) void gemm128x64_wmma(
    const _Float16* __restrict__ A, const _Float16* __restrict__ Bm,
    void* __restrict__ Cp, const float* __restrict__ bias,
    int Kd, int lda, int ldb, int ldc,
    long sAz, long sBz, long sCz, float scale) {
  __shared__ _Float16 sA[2][128 * 40];
  __shared__ _Float16 sB[2][64 * 40];

  const int t    = threadIdx.x;
  const int wave = t >> 5;
  const int lane = t & 31;
  const int half = lane >> 4;
  const int r    = lane & 15;
  const int rp   = wave >> 1;        // wave row-pair: tile rows 2rp, 2rp+1
  const int cp   = (wave & 1) * 2;   // wave col-pair: tile cols cp, cp+1

  const int blockM = blockIdx.x * 128;
  const int blockN = blockIdx.y * 64;
  const long z = blockIdx.z;
  A  += z * sAz;
  Bm += z * sBz;

  v8f acc00 = {}, acc01 = {}, acc10 = {}, acc11 = {};

  // Staging thread mapping
  const int arow = t >> 2;           // 0..63 (and +64 for second A half)
  const int aseg = (t & 3) * 8;      // 0,8,16,24
  const int bk   = t >> 3;           // 0..31  (!TRANSB)
  const int nseg = (t & 7) * 8;      // 0..56  (!TRANSB)

  // Async DMA fill of one K-slab into buffer bufi
  auto async_fill = [&](int bufi, int kk) {
    async_ld_b128(A + (size_t)(blockM + arow) * lda + kk + aseg,
                  &sA[bufi][arow * 40 + aseg]);
    async_ld_b128(A + (size_t)(blockM + 64 + arow) * lda + kk + aseg,
                  &sA[bufi][(arow + 64) * 40 + aseg]);
    if (TRANSB)
      async_ld_b128(Bm + (size_t)(blockN + arow) * ldb + kk + aseg,
                    &sB[bufi][arow * 40 + aseg]);
  };

  v8h rb = {};
  async_fill(0, 0);
  if (!TRANSB) {
    rb = *(const v8h*)(Bm + (size_t)bk * ldb + blockN + nseg);
#pragma unroll
    for (int j = 0; j < 8; ++j) sB[0][(nseg + j) * 40 + bk] = rb[j];
  }

  int buf = 0;
  for (int k0 = 0; k0 < Kd; k0 += 32) {
    const bool has_next = (k0 + 32 < Kd);

    wait_async0();       // our slab's DMA writes to LDS have landed
    __syncthreads();     // everyone's landed; everyone done reading buf^1

    if (has_next) {
      async_fill(buf ^ 1, k0 + 32);          // DMA next slab during compute
      if (!TRANSB) {
        rb = *(const v8h*)(Bm + (size_t)(k0 + 32 + bk) * ldb + blockN + nseg);
        if (k0 + 64 < Kd)
          __builtin_prefetch(Bm + (size_t)(k0 + 64 + bk) * ldb + blockN + nseg,
                             0, 3);
      }
    }

    v16h a0 = load_frag16(&sA[buf][((rp * 2 + 0) * 16 + r) * 40], half);
    v16h a1 = load_frag16(&sA[buf][((rp * 2 + 1) * 16 + r) * 40], half);
    v16h b0 = load_frag16(&sB[buf][((cp + 0) * 16 + r) * 40], half);
    v16h b1 = load_frag16(&sB[buf][((cp + 1) * 16 + r) * 40], half);

    acc00 = __builtin_amdgcn_wmma_f32_16x16x32_f16(false, a0, false, b0,
                                                   (short)0, acc00, false, false);
    acc01 = __builtin_amdgcn_wmma_f32_16x16x32_f16(false, a0, false, b1,
                                                   (short)0, acc01, false, false);
    acc10 = __builtin_amdgcn_wmma_f32_16x16x32_f16(false, a1, false, b0,
                                                   (short)0, acc10, false, false);
    acc11 = __builtin_amdgcn_wmma_f32_16x16x32_f16(false, a1, false, b1,
                                                   (short)0, acc11, false, false);

    if (!TRANSB && has_next) {
      // Transpose-scatter next B slab after the WMMAs (latency hidden)
#pragma unroll
      for (int j = 0; j < 8; ++j) sB[buf ^ 1][(nseg + j) * 40 + bk] = rb[j];
    }
    buf ^= 1;
  }

  // Epilogue. C layout (ISA 7.12.2): VGPR i -> M = i + half*8, N = r.
#pragma unroll
  for (int jr = 0; jr < 2; ++jr) {
#pragma unroll
    for (int jc = 0; jc < 2; ++jc) {
      v8f acc = jr ? (jc ? acc11 : acc10) : (jc ? acc01 : acc00);
      const int gr0 = blockM + (rp * 2 + jr) * 16 + half * 8;
      const int gc  = blockN + (cp + jc) * 16 + r;
      float bv = (MODE == 0 || MODE == 3) ? bias[gc] : 0.0f;
#pragma unroll
      for (int i = 0; i < 8; ++i) {
        size_t idx = (size_t)(gr0 + i) * ldc + gc + (size_t)z * sCz;
        float v = acc[i];
        if (MODE == 0) {
          ((_Float16*)Cp)[idx] = (_Float16)(v + bv);
        } else if (MODE == 1) {
          ((float*)Cp)[idx] = v * scale;
        } else if (MODE == 2) {
          float tz = tanhf(v);
          ((float*)Cp)[idx] = 1.0f / (1.0f + __expf(-tz));
        } else {
          ((float*)Cp)[idx] = v + bv;
        }
      }
    }
  }
}

// ---------------------------------------------------------------------------
// Column softmax (softmax over the QUERY axis q for each (b,k) column).
// Pass 1: online max + sum(exp) down each column (coalesced across k).
// ---------------------------------------------------------------------------
__global__ void colsoftmax_stats_kernel(const float* __restrict__ logits,
                                        float* __restrict__ cmax,
                                        float* __restrict__ csum, int S) {
  int b = blockIdx.y;
  int k = blockIdx.x * blockDim.x + threadIdx.x;
  const float* colp = logits + (size_t)b * S * S + k;
  float m = -3.402823466e38f, s = 0.0f;
  for (int q = 0; q < S; ++q) {
    float v  = colp[(size_t)q * S];
    float nm = fmaxf(m, v);
    s = s * __expf(m - nm) + __expf(v - nm);
    m = nm;
  }
  cmax[b * S + k] = m;
  csum[b * S + k] = s;
}

// Pass 2: normalize and convert to f16 for the W@V WMMA GEMM.
__global__ void colsoftmax_norm_kernel(const float* __restrict__ logits,
                                       const float* __restrict__ cmax,
                                       const float* __restrict__ csum,
                                       _Float16* __restrict__ w, int S,
                                       long total) {
  long i = (long)blockIdx.x * blockDim.x + threadIdx.x;
  if (i >= total) return;
  long row = i / S;              // b*S + q
  int  k   = (int)(i - row * S);
  long b   = row / S;
  float g = __expf(logits[i] - cmax[b * S + k]) / csum[b * S + k];
  w[i] = (_Float16)g;
}

// ---------------------------------------------------------------------------
// Gate mixing + concat into fused [B,S,2D] (f16 for the final WMMA GEMM).
// out1 = a2*self + (1-a2)*conv ; out2 = a1*conv + (1-a1)*self
// ---------------------------------------------------------------------------
__global__ void fuse_gate_kernel(const float* __restrict__ a1,
                                 const float* __restrict__ a2,
                                 const float* __restrict__ xs,
                                 const float* __restrict__ xc,
                                 _Float16* __restrict__ fused, int D,
                                 long total) {
  long i = (long)blockIdx.x * blockDim.x + threadIdx.x;
  if (i >= total) return;
  float g1 = a1[i], g2 = a2[i];
  float s = xs[i], c = xc[i];
  float o1 = g2 * s + (1.0f - g2) * c;
  float o2 = g1 * c + (1.0f - g1) * s;
  long row = i / D;
  int  col = (int)(i - row * D);
  fused[row * (2L * D) + col]     = (_Float16)o1;
  fused[row * (2L * D) + D + col] = (_Float16)o2;
}

// ---------------------------------------------------------------------------
// Host orchestration
// ---------------------------------------------------------------------------
static inline size_t align256(size_t x) { return (x + 255) & ~(size_t)255; }

extern "C" void kernel_launch(void* const* d_in, const int* in_sizes, int n_in,
                              void* d_out, int out_size, void* d_ws,
                              size_t ws_size, hipStream_t stream) {
  (void)in_sizes; (void)n_in; (void)out_size; (void)ws_size;
  const int Bn = BN_C, S = S_C, D = D_C;
  const long M = (long)Bn * S;     // 16384 rows

  const float* self_x = (const float*)d_in[0];
  const float* conv_x = (const float*)d_in[1];
  // d_in[2..13]: Wq1,bq1,Wk1,bk1,Wv1,bv1,Wq2,bq2,Wk2,bk2,Wv2,bv2
  const float* Wf = (const float*)d_in[14];
  const float* bf = (const float*)d_in[15];

  // Workspace carve (with aliasing; ~385 MB total)
  char* p = (char*)d_ws;
  auto carve = [&](size_t bytes) { char* r = p; p += align256(bytes); return r; };
  _Float16* xs_h = (_Float16*)carve((size_t)M * D * 2);
  _Float16* xc_h = (_Float16*)carve((size_t)M * D * 2);
  _Float16* wq_h = (_Float16*)carve((size_t)D * D * 2);
  _Float16* wk_h = (_Float16*)carve((size_t)D * D * 2);
  _Float16* wv_h = (_Float16*)carve((size_t)D * D * 2);
  _Float16* wf_h = (_Float16*)carve((size_t)2 * D * D * 2);
  _Float16* q_h  = (_Float16*)carve((size_t)M * D * 2);
  _Float16* k_h  = (_Float16*)carve((size_t)M * D * 2);
  _Float16* v_h  = (_Float16*)carve((size_t)M * D * 2);
  float*    logits = (float*)carve((size_t)Bn * S * S * 4);  // fits in 192MB L2
  _Float16* w16    = (_Float16*)carve((size_t)Bn * S * S * 2);
  float*    cmax   = (float*)carve((size_t)Bn * S * 4);
  float*    csum   = (float*)carve((size_t)Bn * S * 4);
  float*    a1     = (float*)carve((size_t)M * D * 4);
  float*    a2     = logits;            // logits dead once w16 is built
  _Float16* fused  = w16;               // w16 dead once gate GEMM is done

  auto cdiv = [](long a, long b) { return (unsigned)((a + b - 1) / b); };
  const float inv_scale = 1.0f / sqrtf((float)D);

  // f32 -> f16 inputs + final weight
  cvt_f32_f16_kernel<<<cdiv(M * D, 256), 256, 0, stream>>>(self_x, xs_h, M * D);
  cvt_f32_f16_kernel<<<cdiv(M * D, 256), 256, 0, stream>>>(conv_x, xc_h, M * D);
  cvt_f32_f16_kernel<<<cdiv(2L * D * D, 256), 256, 0, stream>>>(Wf, wf_h, 2L * D * D);

  for (int dir = 0; dir < 2; ++dir) {
    const float* Wq = (const float*)d_in[dir ? 8 : 2];
    const float* bq = (const float*)d_in[dir ? 9 : 3];
    const float* Wk = (const float*)d_in[dir ? 10 : 4];
    const float* bk = (const float*)d_in[dir ? 11 : 5];
    const float* Wv = (const float*)d_in[dir ? 12 : 6];
    const float* bv = (const float*)d_in[dir ? 13 : 7];
    const _Float16* qX = dir ? xc_h : xs_h;   // queries
    const _Float16* kX = dir ? xs_h : xc_h;   // keys/values

    cvt_f32_f16_kernel<<<cdiv((long)D * D, 256), 256, 0, stream>>>(Wq, wq_h, (long)D * D);
    cvt_f32_f16_kernel<<<cdiv((long)D * D, 256), 256, 0, stream>>>(Wk, wk_h, (long)D * D);
    cvt_f32_f16_kernel<<<cdiv((long)D * D, 256), 256, 0, stream>>>(Wv, wv_h, (long)D * D);

    // Projections: [M,768] @ [768,768] + bias -> f16
    dim3 gProj((unsigned)(M / 128), (unsigned)(D / 64), 1);
    gemm128x64_wmma<0, false><<<gProj, 256, 0, stream>>>(
        qX, wq_h, q_h, bq, D, D, D, D, 0, 0, 0, 1.0f);
    gemm128x64_wmma<0, false><<<gProj, 256, 0, stream>>>(
        kX, wk_h, k_h, bk, D, D, D, D, 0, 0, 0, 1.0f);
    gemm128x64_wmma<0, false><<<gProj, 256, 0, stream>>>(
        kX, wv_h, v_h, bv, D, D, D, D, 0, 0, 0, 1.0f);

    // Logits: per batch Q[2048,768] @ K^T -> f32 [2048,2048], scaled
    dim3 gAtt((unsigned)(S / 128), (unsigned)(S / 64), (unsigned)Bn);
    gemm128x64_wmma<1, true><<<gAtt, 256, 0, stream>>>(
        q_h, k_h, logits, nullptr, D, D, D, S,
        (long)S * D, (long)S * D, (long)S * S, inv_scale);

    // Column softmax (over the query axis)
    colsoftmax_stats_kernel<<<dim3((unsigned)(S / 256), (unsigned)Bn), 256, 0,
                              stream>>>(logits, cmax, csum, S);
    long tot = (long)Bn * S * S;
    colsoftmax_norm_kernel<<<cdiv(tot, 256), 256, 0, stream>>>(
        logits, cmax, csum, w16, S, tot);

    // Gate: a = sigmoid(tanh(W @ V)), per batch [2048,2048]@[2048,768]
    dim3 gOut((unsigned)(S / 128), (unsigned)(D / 64), (unsigned)Bn);
    gemm128x64_wmma<2, false><<<gOut, 256, 0, stream>>>(
        w16, v_h, dir ? a2 : a1, nullptr, S, S, D, D,
        (long)S * S, (long)S * D, (long)S * D, 1.0f);
  }

  // Gate mixing + concat
  long totMD = M * (long)D;
  fuse_gate_kernel<<<cdiv(totMD, 256), 256, 0, stream>>>(
      a1, a2, self_x, conv_x, fused, D, totMD);

  // Final: fused [M,1536] @ Wf [1536,768] + bf -> f32 d_out
  gemm128x64_wmma<3, false><<<dim3((unsigned)(M / 128), (unsigned)(D / 64), 1),
                              256, 0, stream>>>(
      fused, wf_h, (float*)d_out, bf, 2 * D, 2 * D, D, D, 0, 0, 0, 1.0f);
}
